// MPNNStack_10969346474866
// MI455X (gfx1250) — matrix operand order repaired and checked by reading, compiled, not run on hardware
//
#include <hip/hip_runtime.h>
#include <hip/hip_bf16.h>

#define NN 50000
#define EE 800000
#define BB 64
#define LL 4
#define ND 128
#define ED 64

typedef _Float16 half16 __attribute__((ext_vector_type(16)));
typedef _Float16 half4v __attribute__((ext_vector_type(4)));
typedef float    float8 __attribute__((ext_vector_type(8)));

__device__ __forceinline__ float gelu_exact(float v) {
    return 0.5f * v * (1.0f + erff(v * 0.70710678118654752f));
}

// A fragment (16x32 f16, row-major source with given stride in halves).
// Lane hl holds row m=hl; halves [k0+8*hi, +8) and [k0+16+8*hi, +8)  (ISA 7.12.2)
__device__ __forceinline__ half16 load_a_frag(const _Float16* base, int stride, int k0,
                                              int hl, int hi) {
    union { float4 f[2]; half16 h; } u;
    const _Float16* p = base + (size_t)hl * stride + k0 + 8 * hi;
    u.f[0] = *(const float4*)p;
    u.f[1] = *(const float4*)(p + 16);
    return u.h;
}

// B fragment (32x16 f16) from transposed weights Wt[n][k] (K halves per row).
// Lane hl holds col n=n0+hl; halves K = [k0+8*hi, +8) and [k0+16+8*hi, +8) (matches A layout)
__device__ __forceinline__ half16 load_b_frag(const _Float16* wt, int K, int n0, int k0,
                                              int hl, int hi) {
    union { float4 f[2]; half16 h; } u;
    const _Float16* p = wt + (size_t)(n0 + hl) * K + k0 + 8 * hi;
    u.f[0] = *(const float4*)p;
    u.f[1] = *(const float4*)(p + 16);
    return u.h;
}

__device__ __forceinline__ float8 wmma_f16(half16 a, half16 b, float8 c) {
    return __builtin_amdgcn_wmma_f32_16x16x32_f16(false, a, false, b, (short)0, c,
                                                  false, false);
}

// ---- weight convert + transpose: W[l][k][n] (f32) -> Wt[l][n][k] (f16) ----
__global__ void convT_kernel(const float* __restrict__ W, _Float16* __restrict__ Wt,
                             int K, int Nout, int total) {
    int i = blockIdx.x * blockDim.x + threadIdx.x;
    if (i >= total) return;
    int n = i % Nout;
    int rest = i / Nout;
    int k = rest % K;
    int l = rest / K;
    Wt[((size_t)l * Nout + n) * K + k] = (_Float16)W[((size_t)l * K + k) * Nout + n];
}

// ---- message MLP + scatter-add: one wave per 16-edge tile, 4 waves/block ----
__global__ __launch_bounds__(128) void msg_kernel(
    const float* __restrict__ x, const int* __restrict__ ei, const float* __restrict__ ea,
    const _Float16* __restrict__ w1t, const float* __restrict__ b1,
    const _Float16* __restrict__ w2t, const float* __restrict__ b2,
    float* __restrict__ aggr)
{
    __shared__ __align__(16) _Float16 sA[4][16 * 320];
    __shared__ __align__(16) _Float16 sM[4][16 * 64];
    __shared__ int sD[4][16];
    __shared__ int sS[4][16];
    const int wv   = threadIdx.x >> 5;
    const int lane = threadIdx.x & 31;
    const int hl   = lane & 15;
    const int hi   = lane >> 4;
    const int e0   = (blockIdx.x * 4 + wv) * 16;

    if (lane < 16) {
        sD[wv][lane] = ei[e0 + lane];        // dst
        sS[wv][lane] = ei[EE + e0 + lane];   // src
    }
    __syncthreads();

    // Gather concat(x[dst], x[src], ea) -> f16 LDS tile [16][320]
    {
        const int r = lane >> 1;
        const int p = lane & 1;
        const int e  = e0 + r;
        const int nd = sD[wv][r];
        const int ns = sS[wv][r];
        _Float16* arow = &sA[wv][r * 320];
        #pragma unroll
        for (int q = 0; q < 40; ++q) {
            const int c = p * 160 + q * 4;
            float4 f;
            if (c < 128)       f = *(const float4*)&x[(size_t)nd * ND + c];
            else if (c < 256)  f = *(const float4*)&x[(size_t)ns * ND + (c - 128)];
            else               f = *(const float4*)&ea[(size_t)e * ED + (c - 256)];
            half4v h = {(_Float16)f.x, (_Float16)f.y, (_Float16)f.z, (_Float16)f.w};
            *(half4v*)(arow + c) = h;
        }
    }
    __syncthreads();

    // GEMM1: [16,320]x[320,64] + bias + gelu -> sM (f16)
    // k-outer / 4-wide accumulator block: each A fragment loaded once, used 4x
    {
        float8 accs[4] = {};
        #pragma unroll
        for (int ks = 0; ks < 10; ++ks) {
            half16 a = load_a_frag(&sA[wv][0], 320, ks * 32, hl, hi);
            #pragma unroll
            for (int nt = 0; nt < 4; ++nt) {
                half16 b = load_b_frag(w1t, 320, nt * 16, ks * 32, hl, hi);
                accs[nt] = wmma_f16(a, b, accs[nt]);
            }
        }
        #pragma unroll
        for (int nt = 0; nt < 4; ++nt) {
            const float bias = b1[nt * 16 + hl];
            #pragma unroll
            for (int j = 0; j < 8; ++j) {
                float v = gelu_exact(accs[nt][j] + bias);
                sM[wv][(j + 8 * hi) * 64 + nt * 16 + hl] = (_Float16)v;
            }
        }
    }
    __syncthreads();

    // GEMM2: [16,64]x[64,64] + bias + gelu -> atomic scatter into aggr[dst]
    {
        float8 accs[4] = {};
        #pragma unroll
        for (int ks = 0; ks < 2; ++ks) {
            half16 a = load_a_frag(&sM[wv][0], 64, ks * 32, hl, hi);
            #pragma unroll
            for (int nt = 0; nt < 4; ++nt) {
                half16 b = load_b_frag(w2t, 64, nt * 16, ks * 32, hl, hi);
                accs[nt] = wmma_f16(a, b, accs[nt]);
            }
        }
        #pragma unroll
        for (int nt = 0; nt < 4; ++nt) {
            const float bias = b2[nt * 16 + hl];
            #pragma unroll
            for (int j = 0; j < 8; ++j) {
                float v = gelu_exact(accs[nt][j] + bias);
                const int row  = j + 8 * hi;
                const int node = sD[wv][row];
                atomicAdd(&aggr[(size_t)node * ED + nt * 16 + hl], v);
            }
        }
    }
}

// ---- update MLP + residual + LayerNorm: one wave per 16-node tile ----
__global__ __launch_bounds__(128) void upd_kernel(
    const float* __restrict__ xin, const float* __restrict__ aggr,
    const _Float16* __restrict__ w1t, const float* __restrict__ b1,
    const _Float16* __restrict__ w2t, const float* __restrict__ b2,
    const float* __restrict__ lng, const float* __restrict__ lnb,
    float* __restrict__ xout, int nTiles)
{
    __shared__ __align__(16) float    sH[4][16 * 128];  // aliases A tile (6144B <= 8192B)
    __shared__ __align__(16) _Float16 sM[4][16 * 128];
    __shared__ float sStat[4][16][2];
    const int wv   = threadIdx.x >> 5;
    const int lane = threadIdx.x & 31;
    const int hl   = lane & 15;
    const int hi   = lane >> 4;
    const int tile = blockIdx.x * 4 + wv;
    const bool active = tile < nTiles;
    const int n0   = tile * 16;
    _Float16* sA = (_Float16*)&sH[wv][0];  // [16][192] f16 view

    if (active) {
        const int r = lane >> 1;
        const int p = lane & 1;
        const int node = n0 + r;
        _Float16* arow = sA + r * 192;
        #pragma unroll
        for (int q = 0; q < 24; ++q) {
            const int c = p * 96 + q * 4;
            float4 f;
            if (c < 128) f = *(const float4*)&xin[(size_t)node * ND + c];
            else         f = *(const float4*)&aggr[(size_t)node * ED + (c - 128)];
            half4v h = {(_Float16)f.x, (_Float16)f.y, (_Float16)f.z, (_Float16)f.w};
            *(half4v*)(arow + c) = h;
        }
    }
    __syncthreads();

    if (active) {
        // GEMM1: [16,192]x[192,128] + gelu -> sM ; two groups of 4 output tiles
        #pragma unroll
        for (int g = 0; g < 2; ++g) {
            float8 accs[4] = {};
            #pragma unroll
            for (int ks = 0; ks < 6; ++ks) {
                half16 a = load_a_frag(sA, 192, ks * 32, hl, hi);
                #pragma unroll
                for (int i = 0; i < 4; ++i) {
                    half16 b = load_b_frag(w1t, 192, (g * 4 + i) * 16, ks * 32, hl, hi);
                    accs[i] = wmma_f16(a, b, accs[i]);
                }
            }
            #pragma unroll
            for (int i = 0; i < 4; ++i) {
                const int nt = g * 4 + i;
                const float bias = b1[nt * 16 + hl];
                #pragma unroll
                for (int j = 0; j < 8; ++j)
                    sM[wv][(j + 8 * hi) * 128 + nt * 16 + hl] =
                        (_Float16)gelu_exact(accs[i][j] + bias);
            }
        }
    }
    __syncthreads();

    if (active) {
        // GEMM2: [16,128]x[128,128] + gelu, residual -> sH (overwrites dead A tile)
        #pragma unroll
        for (int g = 0; g < 2; ++g) {
            float8 accs[4] = {};
            #pragma unroll
            for (int ks = 0; ks < 4; ++ks) {
                half16 a = load_a_frag(&sM[wv][0], 128, ks * 32, hl, hi);
                #pragma unroll
                for (int i = 0; i < 4; ++i) {
                    half16 b = load_b_frag(w2t, 128, (g * 4 + i) * 16, ks * 32, hl, hi);
                    accs[i] = wmma_f16(a, b, accs[i]);
                }
            }
            #pragma unroll
            for (int i = 0; i < 4; ++i) {
                const int nt = g * 4 + i;
                const float bias = b2[nt * 16 + hl];
                const int col = nt * 16 + hl;
                #pragma unroll
                for (int j = 0; j < 8; ++j) {
                    const int row = j + 8 * hi;
                    float u = gelu_exact(accs[i][j] + bias);
                    sH[wv][row * 128 + col] = xin[(size_t)(n0 + row) * ND + col] + u;
                }
            }
        }
    }
    __syncthreads();

    if (active && lane < 16) {
        const int row = lane;
        float mu = 0.f;
        for (int c = 0; c < 128; ++c) mu += sH[wv][row * 128 + c];
        mu *= (1.0f / 128.0f);
        float var = 0.f;
        for (int c = 0; c < 128; ++c) {
            float d = sH[wv][row * 128 + c] - mu;
            var += d * d;
        }
        var *= (1.0f / 128.0f);
        sStat[wv][row][0] = mu;
        sStat[wv][row][1] = rsqrtf(var + 1e-5f);
    }
    __syncthreads();

    if (active) {
        const int row = hl;
        const float mu = sStat[wv][row][0];
        const float rs = sStat[wv][row][1];
        for (int q = 0; q < 64; ++q) {
            const int c = hi * 64 + q;
            float v = (sH[wv][row * 128 + c] - mu) * rs * lng[c] + lnb[c];
            xout[(size_t)(n0 + row) * ND + c] = v;
        }
    }
}

// ---- pooling ----
__global__ void pool_accum_kernel(const float* __restrict__ x, const int* __restrict__ batch,
                                  float* __restrict__ pooled, float* __restrict__ counts) {
    int idx = blockIdx.x * blockDim.x + threadIdx.x;
    if (idx >= NN * 32) return;
    int node = idx >> 5;
    int c0   = (idx & 31) * 4;
    int b    = batch[node];
    const float* row = &x[(size_t)node * ND];
    atomicAdd(&pooled[b * ND + c0 + 0], row[c0 + 0]);
    atomicAdd(&pooled[b * ND + c0 + 1], row[c0 + 1]);
    atomicAdd(&pooled[b * ND + c0 + 2], row[c0 + 2]);
    atomicAdd(&pooled[b * ND + c0 + 3], row[c0 + 3]);
    if ((idx & 31) == 0) atomicAdd(&counts[b], 1.0f);
}

__global__ void pool_fin_kernel(const float* __restrict__ pooled,
                                const float* __restrict__ counts,
                                _Float16* __restrict__ ph) {
    int i = blockIdx.x * blockDim.x + threadIdx.x;
    if (i >= BB * ND) return;
    float cnt = counts[i / ND];
    if (cnt < 1.0f) cnt = 1.0f;
    ph[i] = (_Float16)(pooled[i] / cnt);
}

// ---- final projection: [64,128]x[128,256] + gelu, single block ----
__global__ __launch_bounds__(256) void proj_kernel(const _Float16* __restrict__ ph,
                                                   const _Float16* __restrict__ wt,
                                                   const float* __restrict__ pb,
                                                   float* __restrict__ out) {
    const int wv   = threadIdx.x >> 5;
    const int lane = threadIdx.x & 31;
    const int hl   = lane & 15;
    const int hi   = lane >> 4;
    for (int t = wv; t < 16; t += 8) {
        const int mt = t >> 2;   // 0..3
        const int g  = t & 3;    // group of 4 n-tiles
        float8 accs[4] = {};
        #pragma unroll
        for (int ks = 0; ks < 4; ++ks) {
            half16 a = load_a_frag(ph + (size_t)mt * 16 * 128, 128, ks * 32, hl, hi);
            #pragma unroll
            for (int i = 0; i < 4; ++i) {
                half16 b = load_b_frag(wt, 128, (g * 4 + i) * 16, ks * 32, hl, hi);
                accs[i] = wmma_f16(a, b, accs[i]);
            }
        }
        #pragma unroll
        for (int i = 0; i < 4; ++i) {
            const int n0 = (g * 4 + i) * 16;
            const float bias = pb[n0 + hl];
            #pragma unroll
            for (int j = 0; j < 8; ++j) {
                const int row = mt * 16 + j + 8 * hi;
                out[(size_t)row * 256 + n0 + hl] = gelu_exact(accs[i][j] + bias);
            }
        }
    }
}

extern "C" void kernel_launch(void* const* d_in, const int* in_sizes, int n_in,
                              void* d_out, int out_size, void* d_ws, size_t ws_size,
                              hipStream_t stream) {
    (void)in_sizes; (void)n_in; (void)out_size; (void)ws_size;
    const float* x      = (const float*)d_in[0];
    const int*   ei     = (const int*)d_in[1];
    const float* ea     = (const float*)d_in[2];
    const int*   batch  = (const int*)d_in[3];
    const float* msg_w1 = (const float*)d_in[4];
    const float* msg_b1 = (const float*)d_in[5];
    const float* msg_w2 = (const float*)d_in[6];
    const float* msg_b2 = (const float*)d_in[7];
    const float* upd_w1 = (const float*)d_in[8];
    const float* upd_b1 = (const float*)d_in[9];
    const float* upd_w2 = (const float*)d_in[10];
    const float* upd_b2 = (const float*)d_in[11];
    const float* ln_g   = (const float*)d_in[12];
    const float* ln_b   = (const float*)d_in[13];
    const float* proj_w = (const float*)d_in[14];
    const float* proj_b = (const float*)d_in[15];

    char* p = (char*)d_ws;
    auto alloc = [&](size_t bytes) {
        char* r = p;
        p += (bytes + 255) & ~(size_t)255;
        return r;
    };
    float*    xa     = (float*)alloc((size_t)NN * ND * 4);
    float*    xb     = (float*)alloc((size_t)NN * ND * 4);
    float*    aggr   = (float*)alloc((size_t)NN * ED * 4);
    _Float16* w1t    = (_Float16*)alloc((size_t)LL * 64 * 320 * 2);
    _Float16* w2t    = (_Float16*)alloc((size_t)LL * 64 * 64 * 2);
    _Float16* uw1t   = (_Float16*)alloc((size_t)LL * 128 * 192 * 2);
    _Float16* uw2t   = (_Float16*)alloc((size_t)LL * 128 * 128 * 2);
    _Float16* pjt    = (_Float16*)alloc((size_t)256 * 128 * 2);
    float*    pooled = (float*)alloc((size_t)BB * ND * 4);
    float*    counts = (float*)alloc((size_t)BB * 4);
    _Float16* ph     = (_Float16*)alloc((size_t)BB * ND * 2);

    hipMemcpyAsync(xa, x, (size_t)NN * ND * 4, hipMemcpyDeviceToDevice, stream);
    {
        int tot = LL * 320 * 64;
        convT_kernel<<<(tot + 255) / 256, 256, 0, stream>>>(msg_w1, w1t, 320, 64, tot);
    }
    {
        int tot = LL * 64 * 64;
        convT_kernel<<<(tot + 255) / 256, 256, 0, stream>>>(msg_w2, w2t, 64, 64, tot);
    }
    {
        int tot = LL * 192 * 128;
        convT_kernel<<<(tot + 255) / 256, 256, 0, stream>>>(upd_w1, uw1t, 192, 128, tot);
    }
    {
        int tot = LL * 128 * 128;
        convT_kernel<<<(tot + 255) / 256, 256, 0, stream>>>(upd_w2, uw2t, 128, 128, tot);
    }
    {
        int tot = 128 * 256;
        convT_kernel<<<(tot + 255) / 256, 256, 0, stream>>>(proj_w, pjt, 128, 256, tot);
    }

    float* xin = xa;
    float* xo  = xb;
    const int nTiles = NN / 16;
    for (int l = 0; l < LL; ++l) {
        hipMemsetAsync(aggr, 0, (size_t)NN * ED * 4, stream);
        msg_kernel<<<EE / 64, 128, 0, stream>>>(
            xin, ei, ea,
            w1t + (size_t)l * 64 * 320, msg_b1 + l * 64,
            w2t + (size_t)l * 64 * 64,  msg_b2 + l * 64, aggr);
        upd_kernel<<<(nTiles + 3) / 4, 128, 0, stream>>>(
            xin, aggr,
            uw1t + (size_t)l * 128 * 192, upd_b1 + l * 128,
            uw2t + (size_t)l * 128 * 128, upd_b2 + l * 128,
            ln_g + l * 128, ln_b + l * 128, xo, nTiles);
        float* t = xin; xin = xo; xo = t;
    }
    hipMemsetAsync(pooled, 0, (size_t)BB * ND * 4, stream);
    hipMemsetAsync(counts, 0, (size_t)BB * 4, stream);
    pool_accum_kernel<<<(NN * 32 + 255) / 256, 256, 0, stream>>>(xin, batch, pooled, counts);
    pool_fin_kernel<<<(BB * ND + 255) / 256, 256, 0, stream>>>(pooled, counts, ph);
    proj_kernel<<<1, 256, 0, stream>>>(ph, pjt, proj_b, (float*)d_out);
}
